// GCNClassifier_6176162972265
// MI455X (gfx1250) — matrix operand order repaired and checked by reading, not compiled, not run on hardware
//
#include <hip/hip_runtime.h>
#include <hip/hip_bf16.h>
#include <stdint.h>

// GCN classifier: B=64, S=512, E=256, H=128, V=50000, C=2
#define B_ 64
#define S_ 512
#define E_ 256
#define H_ 128
#define C_ 2

typedef __bf16 v16bf __attribute__((ext_vector_type(16)));
typedef float  v8f   __attribute__((ext_vector_type(8)));

union Frag { int4 q[2]; v16bf v; };

__device__ __forceinline__ unsigned short f2bf(float f) {
  unsigned u = __float_as_uint(f);
  u += 0x7FFFu + ((u >> 16) & 1u);          // round-to-nearest-even
  return (unsigned short)(u >> 16);
}

#define WMMA_BF16(a, b, c) \
  __builtin_amdgcn_wmma_f32_16x16x32_bf16(false, (a), false, (b), (short)0, (c), false, false)

// ---------------- weight cast f32 -> bf16 ----------------
__global__ void k_cast(const float* __restrict__ src, unsigned short* __restrict__ dst, int n) {
  int i = blockIdx.x * blockDim.x + threadIdx.x;
  if (i < n) dst[i] = f2bf(src[i]);
}

// ---------------- embedding lookup: h_f32[B,S,E] and h_bfT[B,E,S] ----------------
__global__ void k_embed(const int* __restrict__ tok, const float* __restrict__ emb,
                        float* __restrict__ hf, unsigned short* __restrict__ hT) {
  int s = blockIdx.x, b = blockIdx.y, e = threadIdx.x;   // blockDim.x == E_
  int t = tok[b * S_ + s];
  float v = emb[(size_t)t * E_ + e];
  hf[((size_t)b * S_ + s) * E_ + e] = v;
  hT[((size_t)b * E_ + e) * S_ + s] = f2bf(v);
}

// ---------------- degree = rowsum(adj)+1, and adj -> bf16 ----------------
__global__ void k_degadj(const float* __restrict__ adj, unsigned short* __restrict__ adjbf,
                         float* __restrict__ deg) {
  __shared__ float red[256];
  int s = blockIdx.x, b = blockIdx.y, t = threadIdx.x;   // 256 threads, row len 512
  const float* row = adj + ((size_t)b * S_ + s) * S_;
  unsigned short* orow = adjbf + ((size_t)b * S_ + s) * S_;
  float a0 = row[t], a1 = row[t + 256];
  orow[t] = f2bf(a0); orow[t + 256] = f2bf(a1);
  red[t] = a0 + a1;
  __syncthreads();
  for (int off = 128; off > 0; off >>= 1) {
    if (t < off) red[t] += red[t + off];
    __syncthreads();
  }
  if (t == 0) deg[b * S_ + s] = red[0] + 1.0f;
}

// epilogue helper: x = acc + h, cast bf16, coalesced across lanes (lane = column n)
__device__ __forceinline__ void bmm_tile_store(const v8f& acc, const float* __restrict__ hf,
                                               unsigned short* __restrict__ xbf,
                                               int b, int m0, int n, int Ein, int half) {
  for (int r = 0; r < 8; ++r) {
    int m = m0 + r + 8 * half;
    size_t idx = ((size_t)b * S_ + m) * Ein + n;
    xbf[idx] = f2bf(acc[r] + hf[idx]);
  }
}

// ---------------- bmm: x = (adj @ h) + h, output bf16 row-major [B,S,Ein] ----------------
// 2x2 register blocking: one wave computes a 32x32 output block (4 WMMA tiles).
// Per k-step: 8 b128 loads feed 4 WMMAs (2 loads/WMMA).
// A-frag: adj_bf rows (two contiguous b128 loads/lane).
// B-frag: h_bfT columns-as-rows (two b128 loads/lane at K = k0+8h and k0+16+8h).
__global__ void k_bmm(const unsigned short* __restrict__ adjbf,
                      const unsigned short* __restrict__ hT,
                      const float* __restrict__ hf,
                      unsigned short* __restrict__ xbf, int Ein) {
  int lane = threadIdx.x & 31, w = threadIdx.x >> 5;
  int half = lane >> 4, ln = lane & 15;
  int j0 = blockIdx.x * 32;                  // feature block (2 tiles)
  int i0 = (blockIdx.y * 8 + w) * 32;        // sequence-row block (2 tiles)
  int b  = blockIdx.z;
  const unsigned short* arow0 = adjbf + ((size_t)b * S_ + (i0 + ln)) * S_ + 8 * half;
  const unsigned short* arow1 = arow0 + (size_t)16 * S_;
  const unsigned short* brow0 = hT + ((size_t)b * Ein + (j0 + ln)) * S_ + 8 * half;
  const unsigned short* brow1 = brow0 + (size_t)16 * S_;
  v8f acc00 = {}, acc01 = {}, acc10 = {}, acc11 = {};
  for (int k0 = 0; k0 < S_; k0 += 32) {
    Frag a0, a1, b0, b1;
    a0.q[0] = *(const int4*)(arow0 + k0);        // K = k0+8h    .. +7
    a0.q[1] = *(const int4*)(arow0 + k0 + 16);   // K = k0+16+8h .. +7
    a1.q[0] = *(const int4*)(arow1 + k0);
    a1.q[1] = *(const int4*)(arow1 + k0 + 16);
    b0.q[0] = *(const int4*)(brow0 + k0);        // K = k0+8h    .. +7
    b0.q[1] = *(const int4*)(brow0 + k0 + 16);   // K = k0+16+8h .. +7
    b1.q[0] = *(const int4*)(brow1 + k0);
    b1.q[1] = *(const int4*)(brow1 + k0 + 16);
    acc00 = WMMA_BF16(a0.v, b0.v, acc00);
    acc01 = WMMA_BF16(a0.v, b1.v, acc01);
    acc10 = WMMA_BF16(a1.v, b0.v, acc10);
    acc11 = WMMA_BF16(a1.v, b1.v, acc11);
  }
  bmm_tile_store(acc00, hf, xbf, b, i0,      j0 + ln,      Ein, half);
  bmm_tile_store(acc01, hf, xbf, b, i0,      j0 + 16 + ln, Ein, half);
  bmm_tile_store(acc10, hf, xbf, b, i0 + 16, j0 + ln,      Ein, half);
  bmm_tile_store(acc11, hf, xbf, b, i0 + 16, j0 + 16 + ln, Ein, half);
}

// ---------------- linear: h = relu((x @ W^T + 2b)/deg); dual store f32 + bf16-transposed ----
// 2x1 row blocking per wave (W fragment reused across 2 WMMAs);
// 8 waves/block cover the 8 output-feature tiles (H=128).
__global__ void k_lin(const unsigned short* __restrict__ xbf,
                      const unsigned short* __restrict__ wbf,
                      const float* __restrict__ bias,
                      const float* __restrict__ deg,
                      float* __restrict__ hf, unsigned short* __restrict__ hT,
                      int Din) {
  int lane = threadIdx.x & 31, w = threadIdx.x >> 5;
  int half = lane >> 4, ln = lane & 15;
  int g0 = blockIdx.x * 32;                  // global row block = b*S + s (2 tiles)
  int j0 = w * 16;                           // output feature tile
  const unsigned short* arow0 = xbf + (size_t)(g0 + ln) * Din + 8 * half;
  const unsigned short* arow1 = arow0 + (size_t)16 * Din;
  const unsigned short* brow  = wbf + (size_t)(j0 + ln) * Din + 8 * half; // W row n = B col n
  v8f acc0 = {}, acc1 = {};
  for (int k0 = 0; k0 < Din; k0 += 32) {
    Frag a0, a1, bb;
    a0.q[0] = *(const int4*)(arow0 + k0);
    a0.q[1] = *(const int4*)(arow0 + k0 + 16);
    a1.q[0] = *(const int4*)(arow1 + k0);
    a1.q[1] = *(const int4*)(arow1 + k0 + 16);
    bb.q[0] = *(const int4*)(brow + k0);
    bb.q[1] = *(const int4*)(brow + k0 + 16);
    acc0 = WMMA_BF16(a0.v, bb.v, acc0);
    acc1 = WMMA_BF16(a1.v, bb.v, acc1);
  }
  float bj = 2.0f * bias[j0 + ln];
  for (int t = 0; t < 2; ++t) {
    const v8f& acc = t ? acc1 : acc0;
    int gb = g0 + 16 * t;
    for (int r = 0; r < 8; ++r) {
      int g = gb + r + 8 * half;
      float v = (acc[r] + bj) / deg[g];
      v = v > 0.0f ? v : 0.0f;
      hf[(size_t)g * H_ + j0 + ln] = v;
      int b = g >> 9, s = g & (S_ - 1);
      hT[((size_t)b * H_ + j0 + ln) * S_ + s] = f2bf(v);
    }
  }
}

// ---------------- max-pool over S + 2-class head ----------------
__global__ void k_pool(const float* __restrict__ hf, const float* __restrict__ Wp,
                       const float* __restrict__ bp, float* __restrict__ out) {
  __shared__ float r0[128], r1[128];
  int b = blockIdx.x, h = threadIdx.x;       // 128 threads
  float m = -3.4e38f;
  const float* base = hf + (size_t)b * S_ * H_ + h;
  for (int s = 0; s < S_; ++s) m = fmaxf(m, base[(size_t)s * H_]);
  r0[h] = m * Wp[h];
  r1[h] = m * Wp[H_ + h];
  __syncthreads();
  for (int off = 64; off > 0; off >>= 1) {
    if (h < off) { r0[h] += r0[h + off]; r1[h] += r1[h + off]; }
    __syncthreads();
  }
  if (h == 0) {
    out[b * 2 + 0] = r0[0] + bp[0];
    out[b * 2 + 1] = r1[0] + bp[1];
  }
}

extern "C" void kernel_launch(void* const* d_in, const int* in_sizes, int n_in,
                              void* d_out, int out_size, void* d_ws, size_t ws_size,
                              hipStream_t stream) {
  const int*   sent = (const int*)d_in[0];
  const float* adj  = (const float*)d_in[1];
  const float* emb  = (const float*)d_in[2];
  const float* W1   = (const float*)d_in[3];
  const float* b1   = (const float*)d_in[4];
  const float* W2   = (const float*)d_in[5];
  const float* b2   = (const float*)d_in[6];
  const float* W3   = (const float*)d_in[7];
  const float* b3   = (const float*)d_in[8];
  const float* Wp   = (const float*)d_in[9];
  const float* bp   = (const float*)d_in[10];
  float* out = (float*)d_out;

  char* ws = (char*)d_ws;
  size_t off = 0;
  auto alloc = [&](size_t bytes) -> char* {
    char* p = ws + off; off += (bytes + 255) & ~(size_t)255; return p;
  };
  unsigned short* adjbf = (unsigned short*)alloc((size_t)B_ * S_ * S_ * 2);  // 33.5 MB
  float*          hf    = (float*)alloc((size_t)B_ * S_ * E_ * 4);           // 33.5 MB
  unsigned short* hT    = (unsigned short*)alloc((size_t)B_ * E_ * S_ * 2);  // 16.8 MB
  unsigned short* xbf   = (unsigned short*)alloc((size_t)B_ * S_ * E_ * 2);  // 16.8 MB
  float*          deg   = (float*)alloc((size_t)B_ * S_ * 4);
  unsigned short* w1bf  = (unsigned short*)alloc((size_t)H_ * E_ * 2);
  unsigned short* w2bf  = (unsigned short*)alloc((size_t)H_ * H_ * 2);
  unsigned short* w3bf  = (unsigned short*)alloc((size_t)H_ * H_ * 2);

  k_cast<<<dim3((H_ * E_ + 255) / 256), 256, 0, stream>>>(W1, w1bf, H_ * E_);
  k_cast<<<dim3((H_ * H_ + 255) / 256), 256, 0, stream>>>(W2, w2bf, H_ * H_);
  k_cast<<<dim3((H_ * H_ + 255) / 256), 256, 0, stream>>>(W3, w3bf, H_ * H_);
  k_embed<<<dim3(S_, B_), 256, 0, stream>>>(sent, emb, hf, hT);
  k_degadj<<<dim3(S_, B_), 256, 0, stream>>>(adj, adjbf, deg);

  const unsigned short* wb[3] = {w1bf, w2bf, w3bf};
  const float* bv[3] = {b1, b2, b3};
  int Ein = E_;
  for (int layer = 0; layer < 3; ++layer) {
    // 32x32 block per wave, 8 waves/block -> 256 rows per block
    k_bmm<<<dim3(Ein / 32, S_ / 256, B_), 256, 0, stream>>>(adjbf, hT, hf, xbf, Ein);
    // 32 rows x 128 features per block
    k_lin<<<dim3((B_ * S_) / 32), 256, 0, stream>>>(xbf, wb[layer], bv[layer], deg, hf, hT, Ein);
    Ein = H_;
  }
  k_pool<<<dim3(B_), 128, 0, stream>>>(hf, Wp, bp, out);
}
